// Block_Ligand_62672162783762
// MI455X (gfx1250) — compile-verified, matrix operand order checked by reading, not run on hardware
//
#include <hip/hip_runtime.h>

#define ND 128
#define ED 64
#define H  8
#define C  16

typedef __attribute__((ext_vector_type(16))) _Float16 v16h;
typedef __attribute__((ext_vector_type(8)))  _Float16 v8h;
typedef __attribute__((ext_vector_type(4)))  _Float16 v4h;
typedef __attribute__((ext_vector_type(8)))  float    v8f;
typedef __attribute__((ext_vector_type(4)))  float    v4f;

// ---------- WMMA helpers (CDNA5 16x16x32 f16, f32 accum) ----------
__device__ __forceinline__ v8f wmma16(v16h a, v16h b, v8f c) {
    return __builtin_amdgcn_wmma_f32_16x16x32_f16(
        /*neg_a=*/false, a, /*neg_b=*/false, b,
        /*c_mod=*/(short)0, c, /*reuse_a=*/false, /*reuse_b=*/false);
}

// Fragment loader for BOTH A (row-major [M][K]) and B stored transposed ([N][K]).
// Per ISA 7.12.2, lane%16 selects the M (or N) index and the 16 K-elements are
// two contiguous 8-half runs at kb and kb+16 -> two ds_load_b128.
__device__ __forceinline__ v16h load_frag(const _Float16* P, int ld, int i0, int k0, int lane) {
    const int i  = i0 + (lane & 15);
    const int kb = k0 + ((lane >> 4) << 3);
    const _Float16* p = P + i * ld + kb;
    union { v16h v; v8h h[2]; } u;
    u.h[0] = *(const v8h*)(p);
    u.h[1] = *(const v8h*)(p + 16);
    return u.v;
}

__device__ __forceinline__ void atomicMaxFloat(float* addr, float val) {
    if (val >= 0.0f) atomicMax((int*)addr, __float_as_int(val));
    else             atomicMin((unsigned int*)addr, __float_as_uint(val));
}

// ============ Kernel 1: time conditioning + LN + Q/K/V GEMMs ============
// 1 wave / block, 16 nodes / block. N % 16 == 0.
__global__ void node_pre_kernel(const float* __restrict__ h,
                                const float* __restrict__ tEmb,
                                const float* __restrict__ Wt, const float* __restrict__ bt,
                                const float* __restrict__ Wq, const float* __restrict__ bq,
                                const float* __restrict__ Wk, const float* __restrict__ bk,
                                const float* __restrict__ Wv, const float* __restrict__ bv,
                                float* __restrict__ qo, float* __restrict__ ko,
                                float* __restrict__ vo) {
    __shared__ _Float16 A[16 * 136];      // silu(time) rows, later LN(h)
    __shared__ _Float16 W[128 * 136];     // weights staged TRANSPOSED: W[n][k], f16
    __shared__ float    HH[16 * 136];     // h tile, then pre-norm rows
    const int lane = threadIdx.x;
    const int r0 = blockIdx.x * 16;

    for (int idx = lane; idx < 16 * ND / 4; idx += 32) {
        const int base = idx * 4;
        const int r = base >> 7, c = base & 127;
        const v4f x = *(const v4f*)(tEmb + (size_t)(r0 + r) * ND + c);
        v4h o;
#pragma unroll
        for (int t = 0; t < 4; ++t) o[t] = (_Float16)(x[t] / (1.0f + expf(-x[t])));
        *(v4h*)&A[r * 136 + c] = o;
        *(v4f*)&HH[r * 136 + c] = *(const v4f*)(h + (size_t)(r0 + r) * ND + c);
    }
    for (int idx = lane; idx < ND * ND / 4; idx += 32) {
        const int base = idx * 4;
        const int r = base >> 7, c = base & 127;   // r = k, c = n
        const v4f w = *(const v4f*)(Wt + base);
#pragma unroll
        for (int t = 0; t < 4; ++t) W[(c + t) * 136 + r] = (_Float16)w[t];
    }
    __syncthreads();

#pragma unroll
    for (int j = 0; j < 8; ++j) {
        v8f acc = {};
#pragma unroll
        for (int kc = 0; kc < 4; ++kc)
            acc = wmma16(load_frag(A, 136, 0, kc * 32, lane),
                         load_frag(W, 136, j * 16, kc * 32, lane), acc);
        const int n = j * 16 + (lane & 15);
#pragma unroll
        for (int rr = 0; rr < 8; ++rr) {
            const int m = rr + ((lane >> 4) << 3);
            HH[m * 136 + n] += acc[rr] + bt[n];    // h + time_proj + bias (LDS RMW)
        }
    }
    __syncthreads();
    if (lane < 16) {   // layernorm over 128
        float mu = 0.f, s2 = 0.f;
        for (int c = 0; c < ND; ++c) mu += HH[lane * 136 + c];
        mu *= (1.f / ND);
        for (int c = 0; c < ND; ++c) { float d = HH[lane * 136 + c] - mu; s2 += d * d; }
        const float rstd = rsqrtf(s2 * (1.f / ND) + 1e-6f);
        for (int c = 0; c < ND; c += 4) {
            v4h o;
#pragma unroll
            for (int t = 0; t < 4; ++t) o[t] = (_Float16)((HH[lane * 136 + c + t] - mu) * rstd);
            *(v4h*)&A[lane * 136 + c] = o;
        }
    }
    __syncthreads();

    const float* Ws[3] = {Wq, Wk, Wv};
    const float* bs[3] = {bq, bk, bv};
    float*       os[3] = {qo, ko, vo};
    for (int t3 = 0; t3 < 3; ++t3) {
        for (int idx = lane; idx < ND * ND / 4; idx += 32) {
            const int base = idx * 4;
            const int r = base >> 7, c = base & 127;
            const v4f w = *(const v4f*)(Ws[t3] + base);
#pragma unroll
            for (int t = 0; t < 4; ++t) W[(c + t) * 136 + r] = (_Float16)w[t];
        }
        __syncthreads();
#pragma unroll
        for (int j = 0; j < 8; ++j) {
            v8f acc = {};
#pragma unroll
            for (int kc = 0; kc < 4; ++kc)
                acc = wmma16(load_frag(A, 136, 0, kc * 32, lane),
                             load_frag(W, 136, j * 16, kc * 32, lane), acc);
            const int n = j * 16 + (lane & 15);
#pragma unroll
            for (int rr = 0; rr < 8; ++rr) {
                const int m = rr + ((lane >> 4) << 3);
                os[t3][(size_t)(r0 + m) * ND + n] = acc[rr] + bs[t3][n];
            }
        }
        __syncthreads();
    }
}

// ============ Kernel 2: RBF + edge embed + LN + e0/e1 GEMMs ============
// 2 waves / block, 32 edges / block. E % 32 == 0. e0/e1 stored f16.
__global__ void edge_kernel(const float* __restrict__ pos,
                            const float* __restrict__ edge_attr,
                            const float* __restrict__ We,  const float* __restrict__ be,
                            const float* __restrict__ We0, const float* __restrict__ We1,
                            const int* __restrict__ eidx, int E,
                            unsigned short* __restrict__ e0o,
                            unsigned short* __restrict__ e1o) {
    __shared__ _Float16 A[32 * 136];      // input tile, later f16 output staging
    __shared__ _Float16 W[128 * 72];      // 9216 halfs: holds [64n][136k] or [128n][72k]
    __shared__ float    EL[32 * 72];
    __shared__ _Float16 EH[32 * 72];
    __shared__ float    dL[32];
    const int tid = threadIdx.x;
    const int lane = tid & 31;
    const int wv = tid >> 5;
    const int eb = blockIdx.x * 32;
    _Float16* e0p = (_Float16*)e0o;
    _Float16* e1p = (_Float16*)e1o;

    if (tid < 32) {
        const int s = eidx[eb + tid];
        const int d = eidx[E + eb + tid];
        const float dx = pos[s * 3 + 0] - pos[d * 3 + 0];
        const float dy = pos[s * 3 + 1] - pos[d * 3 + 1];
        const float dz = pos[s * 3 + 2] - pos[d * 3 + 2];
        dL[tid] = sqrtf(dx * dx + dy * dy + dz * dz);
    }
    __syncthreads();
    const float step  = 15.0f / 63.0f;
    const float coeff = -0.5f / (step * step);
    for (int idx = tid; idx < 32 * 128 / 4; idx += 64) {
        const int base = idx * 4;
        const int r = base >> 7, c = base & 127;
        v4h o;
        if (c < 64) {
            const v4f x = *(const v4f*)(edge_attr + (size_t)(eb + r) * ED + c);
#pragma unroll
            for (int t = 0; t < 4; ++t) o[t] = (_Float16)x[t];
        } else {
            const float d0 = dL[r];
#pragma unroll
            for (int t = 0; t < 4; ++t) {
                const float dd = d0 - step * (float)(c - 64 + t);
                o[t] = (_Float16)expf(coeff * dd * dd);
            }
        }
        *(v4h*)&A[r * 136 + c] = o;
    }
    // W_edge [128k][64n] -> LDS transposed [64n][136k]
    for (int idx = tid; idx < 128 * 64 / 4; idx += 64) {
        const int base = idx * 4;
        const int r = base >> 6, c = base & 63;   // r = k, c = n
        const v4f w = *(const v4f*)(We + base);
#pragma unroll
        for (int t = 0; t < 4; ++t) W[(c + t) * 136 + r] = (_Float16)w[t];
    }
    __syncthreads();

    // e = [edge_attr | rbf] @ W_edge + b_edge  (32 x 64)
#pragma unroll
    for (int j = 0; j < 4; ++j) {
        v8f acc = {};
#pragma unroll
        for (int kc = 0; kc < 4; ++kc)
            acc = wmma16(load_frag(A, 136, wv * 16, kc * 32, lane),
                         load_frag(W, 136, j * 16, kc * 32, lane), acc);
        const int n = j * 16 + (lane & 15);
#pragma unroll
        for (int rr = 0; rr < 8; ++rr) {
            const int m = wv * 16 + rr + ((lane >> 4) << 3);
            EL[m * 72 + n] = acc[rr] + be[n];
        }
    }
    __syncthreads();
    if (tid < 32) {   // layernorm over 64
        float mu = 0.f, s2 = 0.f;
        for (int c = 0; c < ED; ++c) mu += EL[tid * 72 + c];
        mu *= (1.f / ED);
        for (int c = 0; c < ED; ++c) { float d = EL[tid * 72 + c] - mu; s2 += d * d; }
        const float rstd = rsqrtf(s2 * (1.f / ED) + 1e-6f);
        for (int c = 0; c < ED; c += 4) {
            v4h o;
#pragma unroll
            for (int t = 0; t < 4; ++t) o[t] = (_Float16)((EL[tid * 72 + c + t] - mu) * rstd);
            *(v4h*)&EH[tid * 72 + c] = o;
        }
    }
    __syncthreads();

    const float* Wp[2] = {We0, We1};
    _Float16*    op[2] = {e0p, e1p};
    for (int t3 = 0; t3 < 2; ++t3) {
        // W_e [64k][128n] -> LDS transposed [128n][72k]
        for (int idx = tid; idx < 64 * 128 / 4; idx += 64) {
            const int base = idx * 4;
            const int r = base >> 7, c = base & 127;   // r = k, c = n
            const v4f w = *(const v4f*)(Wp[t3] + base);
#pragma unroll
            for (int t = 0; t < 4; ++t) W[(c + t) * 72 + r] = (_Float16)w[t];
        }
        __syncthreads();
#pragma unroll
        for (int j = 0; j < 8; ++j) {
            v8f acc = {};
#pragma unroll
            for (int kc = 0; kc < 2; ++kc)
                acc = wmma16(load_frag(EH, 72, wv * 16, kc * 32, lane),
                             load_frag(W, 72, j * 16, kc * 32, lane), acc);
            const int n = j * 16 + (lane & 15);
#pragma unroll
            for (int rr = 0; rr < 8; ++rr) {
                const int m = wv * 16 + rr + ((lane >> 4) << 3);
                A[m * 136 + n] = (_Float16)acc[rr];    // stage f16 result in LDS
            }
        }
        __syncthreads();
        // wide global stores: 16 bytes per op
        for (int idx = tid; idx < 32 * 128 / 8; idx += 64) {
            const int base = idx * 8;
            const int r = base >> 7, c = base & 127;
            *(v8h*)(op[t3] + (size_t)(eb + r) * ND + c) = *(const v8h*)&A[r * 136 + c];
        }
        __syncthreads();
    }
}

// ============ Kernel 3: init softmax/aggregation state ============
__global__ void init_kernel(float* __restrict__ agg, float* __restrict__ mx,
                            float* __restrict__ den, int N) {
    const int i = blockIdx.x * blockDim.x + threadIdx.x;
    if (i < N * ND) agg[i] = 0.0f;
    if (i < N * H) { mx[i] = -1e30f; den[i] = 0.0f; }
}

// ============ Kernel 4: alpha + segment max ============
__global__ void alpha_kernel(const float* __restrict__ q, const float* __restrict__ k,
                             const unsigned short* __restrict__ e0o,
                             const int* __restrict__ eidx, int E,
                             float* __restrict__ alpha, float* __restrict__ mx) {
    const int gid = blockIdx.x * blockDim.x + threadIdx.x;
    if (gid >= E * H) return;
    const int e = gid >> 3, hh = gid & 7;
    const int s = eidx[e], d = eidx[E + e];
    const v4f* qv = (const v4f*)(q + (size_t)d * ND + hh * C);
    const v4f* kv = (const v4f*)(k + (size_t)s * ND + hh * C);
    const v8h* ev = (const v8h*)((const _Float16*)e0o + (size_t)e * ND + hh * C);
    const v8h e0a = ev[0], e0b = ev[1];
    float acc = 0.f;
#pragma unroll
    for (int cc = 0; cc < 4; ++cc) {
        const v4f qq = qv[cc], kk = kv[cc];
#pragma unroll
        for (int t = 0; t < 4; ++t) {
            const int c = cc * 4 + t;
            const float ee = (float)(c < 8 ? e0a[c & 7] : e0b[c & 7]);
            acc += qq[t] * kk[t] * ee;
        }
    }
    acc *= 0.25f;   // 1/sqrt(16)
    alpha[gid] = acc;
    atomicMaxFloat(&mx[d * H + hh], acc);
}

// ============ Kernel 5: exp + segment sum ============
__global__ void expsum_kernel(const int* __restrict__ eidx, int E,
                              float* __restrict__ alpha, const float* __restrict__ mx,
                              float* __restrict__ den) {
    const int gid = blockIdx.x * blockDim.x + threadIdx.x;
    if (gid >= E * H) return;
    const int e = gid >> 3, hh = gid & 7;
    const int d = eidx[E + e];
    const float ex = expf(alpha[gid] - mx[d * H + hh]);
    alpha[gid] = ex;
    atomicAdd(&den[d * H + hh], ex);
}

// ============ Kernel 6: message scatter ============
__global__ void msg_kernel(const float* __restrict__ v,
                           const unsigned short* __restrict__ e1o,
                           const int* __restrict__ eidx, int E,
                           const float* __restrict__ alpha, const float* __restrict__ den,
                           float* __restrict__ agg) {
    const int gid = blockIdx.x * blockDim.x + threadIdx.x;
    if (gid >= E * H) return;
    const int e = gid >> 3, hh = gid & 7;
    const int s = eidx[e], d = eidx[E + e];
    const float a = alpha[gid] / (den[d * H + hh] + 1e-16f);
    const v4f* vv = (const v4f*)(v + (size_t)s * ND + hh * C);
    const v8h* ev = (const v8h*)((const _Float16*)e1o + (size_t)e * ND + hh * C);
    const v8h e1a = ev[0], e1b = ev[1];
    float* ap = agg + (size_t)d * ND + hh * C;
#pragma unroll
    for (int cc = 0; cc < 4; ++cc) {
        const v4f vx = vv[cc];
#pragma unroll
        for (int t = 0; t < 4; ++t) {
            const int c = cc * 4 + t;
            const float ee = (float)(c < 8 ? e1a[c & 7] : e1b[c & 7]);
            atomicAdd(&ap[c], vx[t] * ee * a);
        }
    }
}

// ============ Kernel 7: residual + LN + FF ============
// 1 wave / block, 16 nodes / block.
__global__ void ff_kernel(const float* __restrict__ h, const float* __restrict__ agg,
                          const float* __restrict__ Wf1, const float* __restrict__ bf1,
                          const float* __restrict__ Wf2, const float* __restrict__ bf2,
                          float* __restrict__ out) {
    __shared__ float    HH[16 * 136];     // hnode, then layernorm(hnode)
    __shared__ _Float16 A[16 * 136];      // hnode f16 (FF input)
    __shared__ _Float16 G[16 * 264];      // silu(ff1) f16; later f32 output staging
    __shared__ _Float16 W[128 * 136];     // weights transposed [n][k]
    const int lane = threadIdx.x;
    const int r0 = blockIdx.x * 16;

    for (int idx = lane; idx < 16 * ND / 4; idx += 32) {
        const int base = idx * 4;
        const int r = base >> 7, c = base & 127;
        const v4f hv = *(const v4f*)(h + (size_t)(r0 + r) * ND + c);
        const v4f av = *(const v4f*)(agg + (size_t)(r0 + r) * ND + c);
        v4h o;
#pragma unroll
        for (int t = 0; t < 4; ++t) {
            const float x = hv[t] + av[t];
            HH[r * 136 + c + t] = x;
            o[t] = (_Float16)x;
        }
        *(v4h*)&A[r * 136 + c] = o;
    }
    __syncthreads();
    if (lane < 16) {   // layernorm hnode in place
        float mu = 0.f, s2 = 0.f;
        for (int c = 0; c < ND; ++c) mu += HH[lane * 136 + c];
        mu *= (1.f / ND);
        for (int c = 0; c < ND; ++c) { float d = HH[lane * 136 + c] - mu; s2 += d * d; }
        const float rstd = rsqrtf(s2 * (1.f / ND) + 1e-6f);
        for (int c = 0; c < ND; ++c) HH[lane * 136 + c] = (HH[lane * 136 + c] - mu) * rstd;
    }
    // FF1 (128 -> 256) in two column halves
    for (int half = 0; half < 2; ++half) {
        for (int idx = lane; idx < ND * ND / 4; idx += 32) {
            const int base = idx * 4;
            const int r = base >> 7, c = base & 127;   // r = k, c = local n
            const v4f w = *(const v4f*)(Wf1 + (size_t)r * 256 + half * 128 + c);
#pragma unroll
            for (int t = 0; t < 4; ++t) W[(c + t) * 136 + r] = (_Float16)w[t];
        }
        __syncthreads();
#pragma unroll
        for (int j = 0; j < 8; ++j) {
            v8f acc = {};
#pragma unroll
            for (int kc = 0; kc < 4; ++kc)
                acc = wmma16(load_frag(A, 136, 0, kc * 32, lane),
                             load_frag(W, 136, j * 16, kc * 32, lane), acc);
            const int n = half * 128 + j * 16 + (lane & 15);
#pragma unroll
            for (int rr = 0; rr < 8; ++rr) {
                const int m = rr + ((lane >> 4) << 3);
                const float x = acc[rr] + bf1[n];
                G[m * 264 + n] = (_Float16)(x / (1.0f + expf(-x)));   // silu
            }
        }
        __syncthreads();
    }
    // FF2 (256 -> 128), K split in two halves, accumulators persist
    v8f acc2[8] = {};
    for (int kh = 0; kh < 2; ++kh) {
        for (int idx = lane; idx < ND * ND / 4; idx += 32) {
            const int base = idx * 4;
            const int r = base >> 7, c = base & 127;   // r = local k, c = n
            const v4f w = *(const v4f*)(Wf2 + (size_t)(kh * 128 + r) * ND + c);
#pragma unroll
            for (int t = 0; t < 4; ++t) W[(c + t) * 136 + r] = (_Float16)w[t];
        }
        __syncthreads();
#pragma unroll
        for (int j = 0; j < 8; ++j) {
#pragma unroll
            for (int kc = 0; kc < 4; ++kc)
                acc2[j] = wmma16(load_frag(G, 264, 0, kh * 128 + kc * 32, lane),
                                 load_frag(W, 136, j * 16, kc * 32, lane), acc2[j]);
        }
        __syncthreads();
    }
    // stage result (f32) into G's storage, then wide store
    float* Gf = (float*)G;                 // 16 x 132 f32 (8448 bytes) fits in G
#pragma unroll
    for (int j = 0; j < 8; ++j) {
        const int n = j * 16 + (lane & 15);
#pragma unroll
        for (int rr = 0; rr < 8; ++rr) {
            const int m = rr + ((lane >> 4) << 3);
            Gf[m * 132 + n] = HH[m * 136 + n] + acc2[j][rr] + bf2[n];
        }
    }
    __syncthreads();
    for (int idx = lane; idx < 16 * ND / 4; idx += 32) {
        const int base = idx * 4;
        const int r = base >> 7, c = base & 127;
        *(v4f*)(out + (size_t)(r0 + r) * ND + c) = *(const v4f*)&Gf[r * 132 + c];
    }
}

extern "C" void kernel_launch(void* const* d_in, const int* in_sizes, int n_in,
                              void* d_out, int out_size, void* d_ws, size_t ws_size,
                              hipStream_t stream) {
    (void)in_sizes; (void)n_in; (void)out_size; (void)ws_size;
    const int N = 50000, E = 800000;
    const float* pos  = (const float*)d_in[0];
    const float* h    = (const float*)d_in[1];
    const float* ea   = (const float*)d_in[2];
    const float* tEmb = (const float*)d_in[3];
    // d_in[4] edge_time_emb: unused by the reference
    const float* We  = (const float*)d_in[5];
    const float* be  = (const float*)d_in[6];
    const float* Wt  = (const float*)d_in[7];
    const float* bt  = (const float*)d_in[8];
    const float* Wq  = (const float*)d_in[9];
    const float* bq  = (const float*)d_in[10];
    const float* Wk  = (const float*)d_in[11];
    const float* bk  = (const float*)d_in[12];
    const float* Wv  = (const float*)d_in[13];
    const float* bv  = (const float*)d_in[14];
    const float* We0 = (const float*)d_in[15];
    const float* We1 = (const float*)d_in[16];
    const float* Wf1 = (const float*)d_in[17];
    const float* bf1 = (const float*)d_in[18];
    const float* Wf2 = (const float*)d_in[19];
    const float* bf2 = (const float*)d_in[20];
    const int* eidx  = (const int*)d_in[21];
    float* out = (float*)d_out;

    char* ws = (char*)d_ws;
    float* q   = (float*)ws; ws += (size_t)N * ND * 4;
    float* k   = (float*)ws; ws += (size_t)N * ND * 4;
    float* v   = (float*)ws; ws += (size_t)N * ND * 4;
    float* agg = (float*)ws; ws += (size_t)N * ND * 4;
    float* mx  = (float*)ws; ws += (size_t)N * H * 4;
    float* den = (float*)ws; ws += (size_t)N * H * 4;
    float* alpha = (float*)ws; ws += (size_t)E * H * 4;
    unsigned short* e0 = (unsigned short*)ws; ws += (size_t)E * ND * 2;
    unsigned short* e1 = (unsigned short*)ws;

    node_pre_kernel<<<N / 16, 32, 0, stream>>>(h, tEmb, Wt, bt, Wq, bq, Wk, bk, Wv, bv, q, k, v);
    edge_kernel<<<E / 32, 64, 0, stream>>>(pos, ea, We, be, We0, We1, eidx, E, e0, e1);
    init_kernel<<<(N * ND + 255) / 256, 256, 0, stream>>>(agg, mx, den, N);
    alpha_kernel<<<(E * H + 255) / 256, 256, 0, stream>>>(q, k, e0, eidx, E, alpha, mx);
    expsum_kernel<<<(E * H + 255) / 256, 256, 0, stream>>>(eidx, E, alpha, mx, den);
    msg_kernel<<<(E * H + 255) / 256, 256, 0, stream>>>(v, e1, eidx, E, alpha, den, agg);
    ff_kernel<<<N / 16, 32, 0, stream>>>(h, agg, Wf1, bf1, Wf2, bf2, out);
}
